// HopeAttention_3032246911477
// MI455X (gfx1250) — compile-verified, matrix operand order checked by reading, not run on hardware
//
#include <hip/hip_runtime.h>
#include <hip/hip_bf16.h>

// ---------------------------------------------------------------------------
// HOPE delta-rule memory, CDNA5 (gfx1250) wave32.
// Big GEMMs: v_wmma_f32_16x16x32_bf16, bf16 inputs (f32 converted in-register
// or bf16 shadows), f32 accumulation. B-operand of the state GEMM staged into
// LDS by the Tensor Data Mover (tensor_load_to_lds + s_wait_tensorcnt),
// double-buffered. State M, reductions, norm clip: f32 VALU.
// ---------------------------------------------------------------------------

#define B_       4
#define S_       4096
#define D_       1024
#define CH_      64
#define NCH_     64
#define MAX_LR_   0.2f
#define MIN_DEC_  0.5f
#define MAX_NRM_  30.0f
#define NRM_EPS_  1e-5f

typedef __attribute__((ext_vector_type(16))) __bf16 v16bf;
typedef __attribute__((ext_vector_type(8)))  float  v8f;

__device__ __forceinline__ float sigmoidf_(float x) { return 1.0f / (1.0f + __expf(-x)); }
__device__ __forceinline__ float siluf_(float x)    { return x * sigmoidf_(x); }

#if defined(__HIP_DEVICE_COMPILE__) && __has_builtin(__builtin_amdgcn_tensor_load_to_lds) && __has_builtin(__builtin_amdgcn_s_wait_tensorcnt)
#define USE_TDM_ 1
#else
#define USE_TDM_ 0
#endif

#if USE_TDM_
typedef __attribute__((ext_vector_type(4))) unsigned int v4u_;
typedef __attribute__((ext_vector_type(8))) int          v8i_;
typedef __attribute__((ext_vector_type(4))) int          v4i_;

// 2-D bf16 tile load via Tensor Data Mover. D# bitfields per ISA 8.3/8.4:
// group0: count=1 | lds_addr | global_addr[56:0] | type=2 ("image").
// group1: data_size=1(2B), tensor_dim0/1, tile_dim0/1, tensor_dim0_stride.
// 6-arg builtin form (this toolchain): (g0, g1, g2, g3, g4, cpol).
__device__ __forceinline__ void tdm_load_2d_bf16(unsigned lds_off, const void* gaddr,
                                                 unsigned tileK, unsigned tileRows,
                                                 unsigned tensorK, unsigned tensorRows,
                                                 unsigned strideK /*elements*/) {
  unsigned long long ga = (unsigned long long)(size_t)gaddr;
  v4u_ g0;
  g0[0] = 1u;                                            // count=1 (valid user D#)
  g0[1] = lds_off;                                       // LDS byte address
  g0[2] = (unsigned)(ga & 0xffffffffu);
  g0[3] = (unsigned)((ga >> 32) & 0x1ffffffu) | (2u << 30);
  v8i_ g1;
  g1[0] = (int)(1u << 16);                               // data_size = 2 bytes
  g1[1] = (int)((tensorK & 0xffffu) << 16);              // tensor_dim0[15:0]
  g1[2] = (int)((tensorK >> 16) | ((tensorRows & 0xffffu) << 16));
  g1[3] = (int)((tensorRows >> 16) | (tileK << 16));     // tile_dim0
  g1[4] = (int)tileRows;                                 // tile_dim1 (tile_dim2=0)
  g1[5] = (int)strideK;                                  // tensor_dim0_stride lo
  g1[6] = 0;
  g1[7] = 0;
  v4i_ z4 = {0, 0, 0, 0};
  v8i_ z8 = {0, 0, 0, 0, 0, 0, 0, 0};
  __builtin_amdgcn_tensor_load_to_lds(g0, g1, z4, z4, z8, 0);
}
#endif

// --- WMMA fragment loader (ISA 7.12.2 16-bit layout) -----------------------
// lanes 0-15: row=lane, K={0..7}∪{16..23}; lanes 16-31: row=lane-16,
// K={8..15}∪{24..31}. Source f32 (cvt in-register) or bf16; generic pointers
// so LDS-resident operands work.
template <typename T>
__device__ __forceinline__ v16bf load_frag(const T* base, int ld) {
  int lane = threadIdx.x & 31;
  int r    = lane & 15;
  int kb   = (lane < 16) ? 0 : 8;
  const T* p = base + (size_t)r * ld + kb;
  v16bf f;
#pragma unroll
  for (int e = 0; e < 8; ++e) {
    f[e]     = (__bf16)p[e];
    f[e + 8] = (__bf16)p[e + 16];
  }
  return f;
}

// One A fragment reused across NF B fragments (16 x NF*16 tile, K mult of 32).
template <int NF, typename TA, typename TB>
__device__ __forceinline__ void wmma_gemm_row(const TA* A, int lda,
                                              const TB* Bt, int ldb, int K,
                                              v8f* acc) {
  for (int k = 0; k < K; k += 32) {
    v16bf a = load_frag(A + k, lda);
#pragma unroll
    for (int f = 0; f < NF; ++f) {
      v16bf b = load_frag(Bt + (size_t)(f * 16) * ldb + k, ldb);
      acc[f] = __builtin_amdgcn_wmma_f32_16x16x32_bf16(false, a, false, b,
                                                       (short)0, acc[f], false, false);
    }
  }
}

__device__ __forceinline__ void zero_acc4(v8f* acc) {
  v8f z = {0.f, 0.f, 0.f, 0.f, 0.f, 0.f, 0.f, 0.f};
#pragma unroll
  for (int f = 0; f < 4; ++f) acc[f] = z;
}

// --- small utility kernels -------------------------------------------------
__global__ void k_f32_to_bf16(const float* __restrict__ in,
                              __bf16* __restrict__ out, size_t n) {
  size_t i = (size_t)blockIdx.x * blockDim.x + threadIdx.x;
  if (i < n) out[i] = (__bf16)in[i];
}

__global__ void k_init_M(const float* __restrict__ Mi,
                         float* __restrict__ Mf, __bf16* __restrict__ Mb) {
  size_t dd = (size_t)D_ * D_;
  size_t i  = (size_t)blockIdx.x * blockDim.x + threadIdx.x;
  if (i < (size_t)B_ * dd) {
    float v = Mi[i % dd];
    Mf[i] = v;
    Mb[i] = (__bf16)v;
  }
}

// --- per-chunk scalar stats: eta, alpha, k_mean, gate ----------------------
__global__ void k_stats(const float* __restrict__ x,
                        const float* __restrict__ eta_w,  const float* __restrict__ eta_b,
                        const float* __restrict__ alpha_w,const float* __restrict__ alpha_b,
                        const float* __restrict__ gate_w, const float* __restrict__ gate_b,
                        int c,
                        float* __restrict__ kmean, float* __restrict__ eta_o,
                        float* __restrict__ alpha_o, float* __restrict__ gate_o) {
  __shared__ float s_km[D_];
  __shared__ float r0[256], r1[256], r2[256];
  __shared__ float s_acc[3];
  int b = blockIdx.x, tid = threadIdx.x;
  for (int i = tid; i < D_; i += 256) s_km[i] = 0.f;
  if (tid == 0) { s_acc[0] = 0.f; s_acc[1] = 0.f; }
  __syncthreads();
  const float* xb = x + ((size_t)b * S_ + (size_t)c * CH_) * D_;
  for (int t = 0; t < CH_; ++t) {
    const float* row = xb + (size_t)t * D_;
    float de = 0.f, da = 0.f, ss = 0.f;
    for (int i = tid; i < D_; i += 256) {
      float v = row[i];
      de += v * eta_w[i];
      da += v * alpha_w[i];
      ss += v * v;
    }
    r0[tid] = de; r1[tid] = da; r2[tid] = ss;
    __syncthreads();
    for (int s = 128; s > 0; s >>= 1) {
      if (tid < s) { r0[tid] += r0[tid+s]; r1[tid] += r1[tid+s]; r2[tid] += r2[tid+s]; }
      __syncthreads();
    }
    if (tid == 0) {
      s_acc[0] += sigmoidf_(r0[0] + eta_b[0]) * MAX_LR_;
      s_acc[1] += MIN_DEC_ + sigmoidf_(r1[0] + alpha_b[0]) * (1.0f - MIN_DEC_);
      s_acc[2]  = fmaxf(sqrtf(r2[0]), NRM_EPS_);
    }
    __syncthreads();
    float inv = 1.0f / s_acc[2];
    for (int i = tid; i < D_; i += 256) s_km[i] += row[i] * inv;
    __syncthreads();
  }
  float dg = 0.f;
  for (int i = tid; i < D_; i += 256) {
    float km = s_km[i] * (1.0f / CH_);
    kmean[(size_t)b * D_ + i] = km;
    dg += km * gate_w[i];
  }
  r0[tid] = dg;
  __syncthreads();
  for (int s = 128; s > 0; s >>= 1) { if (tid < s) r0[tid] += r0[tid+s]; __syncthreads(); }
  if (tid == 0) {
    eta_o[b]   = s_acc[0] * (1.0f / CH_);
    alpha_o[b] = s_acc[1] * (1.0f / CH_);
    gate_o[b]  = sigmoidf_(r0[0] + gate_b[0]);
  }
}

// --- outputs = chunk @ M^T; B tile of M staged to LDS via TDM --------------
__global__ void __launch_bounds__(128)
k_outputs(const float* __restrict__ x, const __bf16* __restrict__ Mb,
          int c, float* __restrict__ outs /*d_out*/) {
  __shared__ __bf16 sB[2][64 * 32];   // double-buffered B tile, 8 KB
  int b = blockIdx.y, nt = blockIdx.x;
  int wid = threadIdx.x >> 5, lane = threadIdx.x & 31;
  int rb = (lane < 16) ? 0 : 8, cl = lane & 15;
  const float*  A     = x  + ((size_t)b * S_ + (size_t)c * CH_ + wid * 16) * D_;
  const __bf16* Bbase = Mb + (size_t)b * D_ * D_ + (size_t)(nt * 64) * D_;
  v8f acc[4];
  zero_acc4(acc);
  const int NKS = D_ / 32;
#if USE_TDM_
  if (wid == 0)
    tdm_load_2d_bf16((unsigned)(size_t)&sB[0][0], Bbase, 32, 64, D_, D_, D_);
#endif
  for (int ks = 0; ks < NKS; ++ks) {
    int cur = ks & 1;
#if USE_TDM_
    __syncthreads();                     // prior compute done -> buffer free
    if (wid == 0) {
      if (ks + 1 < NKS) {
        tdm_load_2d_bf16((unsigned)(size_t)&sB[cur ^ 1][0],
                         Bbase + (ks + 1) * 32, 32, 64, D_, D_, D_);
        __builtin_amdgcn_s_wait_tensorcnt(1);   // current tile done, next in flight
      } else {
        __builtin_amdgcn_s_wait_tensorcnt(0);   // drain last tile
      }
    }
    __syncthreads();                     // publish sB[cur] to all waves
#else
    cur = 0;
    __syncthreads();
    for (int i = threadIdx.x; i < 64 * 32; i += 128)
      sB[0][i] = Bbase[(size_t)(i >> 5) * D_ + ks * 32 + (i & 31)];
    __syncthreads();
#endif
    v16bf a = load_frag(A + ks * 32, D_);
#pragma unroll
    for (int f = 0; f < 4; ++f) {
      v16bf bb = load_frag(&sB[cur][f * 16 * 32], 32);
      acc[f] = __builtin_amdgcn_wmma_f32_16x16x32_bf16(false, a, false, bb,
                                                       (short)0, acc[f], false, false);
    }
  }
#pragma unroll
  for (int f = 0; f < 4; ++f)
#pragma unroll
    for (int v = 0; v < 8; ++v) {
      int r = wid * 16 + rb + v;
      outs[((size_t)b * S_ + (size_t)c * CH_ + r) * D_ + nt * 64 + f * 16 + cl] = acc[f][v];
    }
}

// --- h = silu(outputs @ W1^T) ---------------------------------------------
__global__ void __launch_bounds__(128)
k_hgen(const float* __restrict__ outs, const __bf16* __restrict__ W1b,
       int c, __bf16* __restrict__ H) {
  int b = blockIdx.y, nt = blockIdx.x;
  int wid = threadIdx.x >> 5, lane = threadIdx.x & 31;
  int rb = (lane < 16) ? 0 : 8, cl = lane & 15;
  const float* A = outs + ((size_t)b * S_ + (size_t)c * CH_ + wid * 16) * D_;
  v8f acc[4];
  zero_acc4(acc);
  wmma_gemm_row<4>(A, D_, W1b + (size_t)(nt * 64) * D_, D_, D_, acc);
#pragma unroll
  for (int f = 0; f < 4; ++f)
#pragma unroll
    for (int v = 0; v < 8; ++v) {
      int r = wid * 16 + rb + v;
      H[((size_t)b * CH_ + r) * D_ + nt * 64 + f * 16 + cl] = (__bf16)siluf_(acc[f][v]);
    }
}

// --- v_target = mean_t( h @ W2^T + outputs ) -------------------------------
__global__ void __launch_bounds__(128)
k_vgen(const __bf16* __restrict__ H, const __bf16* __restrict__ W2b,
       const float* __restrict__ outs, int c, float* __restrict__ vtgt) {
  __shared__ float s_tile[64][65];
  int b = blockIdx.y, nt = blockIdx.x;
  int wid = threadIdx.x >> 5, lane = threadIdx.x & 31;
  int rb = (lane < 16) ? 0 : 8, cl = lane & 15;
  const __bf16* A = H + ((size_t)b * CH_ + wid * 16) * D_;
  v8f acc[4];
  zero_acc4(acc);
  wmma_gemm_row<4>(A, D_, W2b + (size_t)(nt * 64) * D_, D_, D_, acc);
#pragma unroll
  for (int f = 0; f < 4; ++f)
#pragma unroll
    for (int v = 0; v < 8; ++v) {
      int r = wid * 16 + rb + v;
      size_t yi = ((size_t)b * S_ + (size_t)c * CH_ + r) * D_ + nt * 64 + f * 16 + cl;
      s_tile[r][f * 16 + cl] = acc[f][v] + outs[yi];
    }
  __syncthreads();
  if (threadIdx.x < 64) {
    float s = 0.f;
    for (int r = 0; r < 64; ++r) s += s_tile[r][threadIdx.x];
    vtgt[(size_t)b * D_ + nt * 64 + threadIdx.x] = s * (1.0f / CH_);
  }
}

// --- delta-rule M update (rows r0..r0+31), deterministic partial norms -----
__global__ void __launch_bounds__(256)
k_mupdate(float* __restrict__ Mf, const float* __restrict__ kmean,
          const float* __restrict__ vtgt, const float* __restrict__ eta_o,
          const float* __restrict__ alpha_o, const float* __restrict__ gate_o,
          float* __restrict__ nsq_part) {
  __shared__ float s_km[D_];
  __shared__ float red[256];
  __shared__ float s_err;
  int b = blockIdx.y, tid = threadIdx.x;
  int r0 = blockIdx.x * 32;
  for (int i = tid; i < D_; i += 256) s_km[i] = kmean[(size_t)b * D_ + i];
  __syncthreads();
  float eta = eta_o[b], alpha = alpha_o[b], g = gate_o[b];
  float* M = Mf + (size_t)b * D_ * D_;
  float ssl = 0.f;
  for (int o = r0; o < r0 + 32; ++o) {
    float* row = M + (size_t)o * D_;
    float d = 0.f;
    for (int i = tid; i < D_; i += 256) d += row[i] * s_km[i];
    red[tid] = d;
    __syncthreads();
    for (int s = 128; s > 0; s >>= 1) { if (tid < s) red[tid] += red[tid+s]; __syncthreads(); }
    if (tid == 0) s_err = vtgt[(size_t)b * D_ + o] - red[0];
    __syncthreads();
    float coef = eta * s_err;
    for (int i = tid; i < D_; i += 256) {
      float mold = row[i];
      float mn = alpha * mold + coef * s_km[i];
      mn = g * mn + (1.0f - g) * mold;
      row[i] = mn;
      ssl += mn * mn;
    }
    __syncthreads();
  }
  red[tid] = ssl;
  __syncthreads();
  for (int s = 128; s > 0; s >>= 1) { if (tid < s) red[tid] += red[tid+s]; __syncthreads(); }
  if (tid == 0) nsq_part[b * 32 + blockIdx.x] = red[0];
}

// --- Frobenius clip + refresh bf16 M shadow --------------------------------
__global__ void __launch_bounds__(256)
k_mscale(float* __restrict__ Mf, __bf16* __restrict__ Mb,
         const float* __restrict__ nsq_part) {
  int b = blockIdx.y;
  float ss = 0.f;
#pragma unroll
  for (int i = 0; i < 32; ++i) ss += nsq_part[b * 32 + i];
  float scale = fminf(MAX_NRM_ / (sqrtf(ss) + 1e-6f), 1.0f);
  size_t base = (size_t)b * D_ * D_ + (size_t)blockIdx.x * 1024;
#pragma unroll
  for (int kk = 0; kk < 4; ++kk) {
    size_t i = base + threadIdx.x + kk * 256;
    float v = Mf[i] * scale;
    Mf[i] = v;
    Mb[i] = (__bf16)v;
  }
}

__global__ void k_copyM(const float* __restrict__ Mf, float* __restrict__ out) {
  size_t i = (size_t)blockIdx.x * 256 + threadIdx.x;
  out[i] = Mf[i];
}

// --- final out = value_gen(outs), fused per 16-row block, h in LDS ---------
__global__ void __launch_bounds__(128)
k_finalvg(const __bf16* __restrict__ W1b, const __bf16* __restrict__ W2b,
          float* __restrict__ io /*d_out, in-place*/) {
  __shared__ __bf16 s_h[16 * D_];  // 32 KB
  size_t row0 = (size_t)blockIdx.x * 16;
  int wid = threadIdx.x >> 5, lane = threadIdx.x & 31;
  int rb = (lane < 16) ? 0 : 8, cl = lane & 15;
  const float* A = io + row0 * D_;
#pragma unroll
  for (int g = 0; g < 4; ++g) {
    int ntb = wid * 16 + g * 4;          // this wave's 4 column tiles
    v8f acc[4];
    zero_acc4(acc);
    wmma_gemm_row<4>(A, D_, W1b + (size_t)(ntb * 16) * D_, D_, D_, acc);
#pragma unroll
    for (int f = 0; f < 4; ++f)
#pragma unroll
      for (int v = 0; v < 8; ++v)
        s_h[(rb + v) * D_ + (ntb + f) * 16 + cl] = (__bf16)siluf_(acc[f][v]);
  }
  __syncthreads();
#pragma unroll
  for (int g = 0; g < 4; ++g) {
    int ntb = wid * 16 + g * 4;
    v8f acc[4];
    zero_acc4(acc);
    wmma_gemm_row<4>((const __bf16*)s_h, D_, W2b + (size_t)(ntb * 16) * D_, D_, D_, acc);
#pragma unroll
    for (int f = 0; f < 4; ++f)
#pragma unroll
      for (int v = 0; v < 8; ++v) {
        size_t idx = (row0 + rb + v) * D_ + (ntb + f) * 16 + cl;
        io[idx] = acc[f][v] + io[idx];
      }
  }
}

// ---------------------------------------------------------------------------
extern "C" void kernel_launch(void* const* d_in, const int* in_sizes, int n_in,
                              void* d_out, int out_size, void* d_ws, size_t ws_size,
                              hipStream_t stream) {
  (void)in_sizes; (void)n_in; (void)out_size; (void)ws_size;
  const float* x       = (const float*)d_in[0];
  const float* M_init  = (const float*)d_in[1];
  const float* eta_w   = (const float*)d_in[2];
  const float* eta_b   = (const float*)d_in[3];
  const float* alpha_w = (const float*)d_in[4];
  const float* alpha_b = (const float*)d_in[5];
  const float* gate_w  = (const float*)d_in[6];
  const float* gate_b  = (const float*)d_in[7];
  const float* vg_w1   = (const float*)d_in[8];
  const float* vg_w2   = (const float*)d_in[9];
  float* out = (float*)d_out;               // [0, B*S*D): final_out; then M_final

  char* ws = (char*)d_ws;
  size_t off = 0;
  auto alloc = [&](size_t bytes) { size_t p = off; off += (bytes + 255) & ~(size_t)255; return p; };
  float*  Mf    = (float*) (ws + alloc((size_t)B_ * D_ * D_ * 4));  // f32 state
  __bf16* Mb    = (__bf16*)(ws + alloc((size_t)B_ * D_ * D_ * 2));  // bf16 shadow
  __bf16* W1b   = (__bf16*)(ws + alloc((size_t)D_ * D_ * 2));
  __bf16* W2b   = (__bf16*)(ws + alloc((size_t)D_ * D_ * 2));
  __bf16* Hb    = (__bf16*)(ws + alloc((size_t)B_ * CH_ * D_ * 2));
  float*  kmean = (float*) (ws + alloc((size_t)B_ * D_ * 4));
  float*  vtgt  = (float*) (ws + alloc((size_t)B_ * D_ * 4));
  float*  eta_o = (float*) (ws + alloc(B_ * 4));
  float*  alp_o = (float*) (ws + alloc(B_ * 4));
  float*  gat_o = (float*) (ws + alloc(B_ * 4));
  float*  nsq   = (float*) (ws + alloc(B_ * 32 * 4));

  size_t nw = (size_t)D_ * D_;
  k_f32_to_bf16<<<dim3((unsigned)((nw + 255) / 256)), 256, 0, stream>>>(vg_w1, W1b, nw);
  k_f32_to_bf16<<<dim3((unsigned)((nw + 255) / 256)), 256, 0, stream>>>(vg_w2, W2b, nw);
  k_init_M<<<dim3((unsigned)((B_ * nw + 255) / 256)), 256, 0, stream>>>(M_init, Mf, Mb);

  for (int c = 0; c < NCH_; ++c) {
    k_stats  <<<dim3(B_),       256, 0, stream>>>(x, eta_w, eta_b, alpha_w, alpha_b,
                                                  gate_w, gate_b, c, kmean, eta_o, alp_o, gat_o);
    k_outputs<<<dim3(16, B_),   128, 0, stream>>>(x, Mb, c, out);
    k_hgen   <<<dim3(16, B_),   128, 0, stream>>>(out, W1b, c, Hb);
    k_vgen   <<<dim3(16, B_),   128, 0, stream>>>(Hb, W2b, out, c, vtgt);
    k_mupdate<<<dim3(32, B_),   256, 0, stream>>>(Mf, kmean, vtgt, eta_o, alp_o, gat_o, nsq);
    k_mscale <<<dim3(1024, B_), 256, 0, stream>>>(Mf, Mb, nsq);
  }
  k_copyM  <<<dim3((unsigned)(B_ * nw / 256)), 256, 0, stream>>>(Mf, out + (size_t)B_ * S_ * D_);
  k_finalvg<<<dim3(B_ * S_ / 16),              128, 0, stream>>>(W1b, W2b, out);
}
